// Smartpool_69990787056254
// MI455X (gfx1250) — matrix operand order repaired and checked by reading, not compiled, not run on hardware
//
#include <hip/hip_runtime.h>
#include <hip/hip_bf16.h>

typedef __attribute__((ext_vector_type(16))) _Float16 v16h;
typedef __attribute__((ext_vector_type(8)))  _Float16 v8h;
typedef __attribute__((ext_vector_type(8)))  float    v8f;

#define B_    8
#define T_    4096
#define C_    512
#define D_    2048
#define BT_   (B_ * T_)
#define TNEW_ (T_ / 2)

#define USE_TDM 1   // Tensor Data Mover staging (set 0 to fall back to global loads)

// ---------------------------------------------------------------------------
// fp32 -> f16 convert (elementwise)
// ---------------------------------------------------------------------------
__global__ void k_convert_f16(const float* __restrict__ in, _Float16* __restrict__ out, int n) {
  int i = blockIdx.x * blockDim.x + threadIdx.x;
  if (i < n) out[i] = (_Float16)in[i];
}

// fp32 [K,N] -> f16 [N,K]  (transpose so GEMM B-tiles are K-contiguous)
__global__ void k_convert_transpose_f16(const float* __restrict__ in, _Float16* __restrict__ out,
                                        int K, int N) {
  int i = blockIdx.x * blockDim.x + threadIdx.x;  // linear over input (coalesced read)
  if (i < K * N) {
    int k = i / N, n = i % N;
    out[(size_t)n * K + k] = (_Float16)in[i];
  }
}

// ---------------------------------------------------------------------------
// TDM: issue tensor_load_to_lds for a 2-D f16 tile [128 rows x 32 elems],
// row stride K, into LDS with native padding (16 DWORDs data + 4 DWORDs pad
// per row -> 40-half LDS row stride). Descriptor per CDNA5 ISA.
// Wave-level op: EXEC ignored, tracked with TENSORcnt.
// ---------------------------------------------------------------------------
#if USE_TDM
__device__ __forceinline__ void tdm_load_tile_f16(
    const _Float16* gtile,      // global address of tile start (within tensor)
    unsigned lds_addr,          // LDS byte address of tile destination
    unsigned K_elems,           // tensor_dim0 == tensor_dim0_stride (row length)
    unsigned tensor_rows)       // tensor_dim1 (rows in whole tensor, for OOB)
{
  typedef __attribute__((ext_vector_type(4))) unsigned u32x4;
  typedef __attribute__((ext_vector_type(8))) unsigned u32x8;
  const unsigned long long ga = (unsigned long long)(uintptr_t)gtile;
  u32x4 g0;
  g0[0] = 1u;                                  // count=1 (valid), no gather
  g0[1] = lds_addr;                            // lds_addr [63:32]
  g0[2] = (unsigned)ga;                        // global_addr low  [95:64]
  g0[3] = (unsigned)(ga >> 32) | (2u << 30);   // global_addr high + type=2 [127:126]
  u32x8 g1;
  g1[0] = (1u << 16)                           // data_size = 2 bytes
        | (1u << 20)                           // pad_enable
        | (3u << 22)                           // pad_interval: 16 DWORDs (64B row)
        | (3u << 25);                          // pad_amount:    4 DWORDs (8 halves)
  g1[1] = (K_elems & 0xFFFFu) << 16;           // tensor_dim0[15:0]   @ bits[63:48]
  g1[2] = (K_elems >> 16)                      // tensor_dim0[31:16]  @ bits[79:64]
        | ((tensor_rows & 0xFFFFu) << 16);     // tensor_dim1[15:0]   @ bits[111:96]... [95:80]
  g1[3] = (tensor_rows >> 16)                  // tensor_dim1[31:16]
        | (32u << 16);                         // tile_dim0 = 32      @ bits[127:112]
  g1[4] = 128u;                                // tile_dim1 = 128, tile_dim2 = 0
  g1[5] = K_elems;                             // tensor_dim0_stride[31:0]
  g1[6] = 0u;                                  // stride[47:32]=0, dim1_stride=0 (2D)
  g1[7] = 0u;
  asm volatile("tensor_load_to_lds %0, %1" :: "s"(g0), "s"(g1) : "memory");
}
#endif

// ---------------------------------------------------------------------------
// Tiled f16 WMMA GEMM: Out[M,N] = act(A[M,K] * Bt[N,K]^T + bias)
// block = 256 threads = 8 waves, tile 128x128, K-step 32, double-buffered LDS.
// wave (wm=wave&3, wn=wave>>2) owns a 32x64 sub-tile: 2x4 frags of 16x16.
// ---------------------------------------------------------------------------
#define TM 128
#define TN 128
#define TK 32
#define LDT 40                 // padded LDS row stride in halves (80B)
#define TILEH (TM * LDT)       // halves per tile buffer

__global__ __launch_bounds__(256) void k_gemm_f16(
    const _Float16* __restrict__ A,   // [M,K]
    const _Float16* __restrict__ Bt,  // [N,K]
    const float*    __restrict__ bias,// [N]
    _Float16*       __restrict__ Out, // [M,N]
    int M, int N, int K, int apply_gelu)
{
  __shared__ _Float16 sA[2 * TILEH];
  __shared__ _Float16 sB[2 * TILEH];

  const int tid  = threadIdx.x;
  const int wave = tid >> 5;          // wave32
  const int lane = tid & 31;
  const int lm   = lane & 15;
  const int lh   = lane >> 4;
  const int wm   = wave & 3;          // m offset = wm*32
  const int wn   = wave >> 2;         // n offset = wn*64

  const int bm = blockIdx.y * TM;
  const int bn = blockIdx.x * TN;

  const _Float16* gAblk = A  + (size_t)bm * K;
  const _Float16* gBblk = Bt + (size_t)bn * K;

  union Frag { v16h v; v8h h[2]; };
  v8f acc[2][4];
  #pragma unroll
  for (int i = 0; i < 2; ++i)
    #pragma unroll
    for (int j = 0; j < 4; ++j)
      #pragma unroll
      for (int r = 0; r < 8; ++r) acc[i][j][r] = 0.0f;

#if USE_TDM
  const bool issuer = (wave == 0);
  const unsigned ldsA = (unsigned)(uintptr_t)&sA[0];
  const unsigned ldsB = (unsigned)(uintptr_t)&sB[0];
  const unsigned tileBytes = TILEH * 2;

  if (issuer) {                         // prologue: tile 0 -> buffer 0
    tdm_load_tile_f16(gAblk, ldsA, (unsigned)K, (unsigned)M);
    tdm_load_tile_f16(gBblk, ldsB, (unsigned)K, (unsigned)N);
  }
  int pb = 0;
  for (int k0 = 0; k0 < K; k0 += TK, pb ^= 1) {
    if (issuer) __builtin_amdgcn_s_wait_tensorcnt(0);  // current tiles landed
    __syncthreads();                                   // publish to all waves
    if (issuer && (k0 + TK < K)) {                     // DMA next pair under compute
      tdm_load_tile_f16(gAblk + (k0 + TK), ldsA + (pb ^ 1) * tileBytes, (unsigned)K, (unsigned)M);
      tdm_load_tile_f16(gBblk + (k0 + TK), ldsB + (pb ^ 1) * tileBytes, (unsigned)K, (unsigned)N);
    }
    const _Float16* cA = &sA[pb * TILEH];
    const _Float16* cB = &sB[pb * TILEH];
#else
  const int lrow = tid >> 1;
  const int lcol = (tid & 1) * 16;
  const _Float16* gA = gAblk + (size_t)lrow * K + lcol;
  const _Float16* gB = gBblk + (size_t)lrow * K + lcol;
  for (int k0 = 0; k0 < K; k0 += TK) {
    v8h a0 = *(const v8h*)(gA + k0);
    v8h a1 = *(const v8h*)(gA + k0 + 8);
    v8h b0 = *(const v8h*)(gB + k0);
    v8h b1 = *(const v8h*)(gB + k0 + 8);
    *(v8h*)&sA[lrow * LDT + lcol]     = a0;
    *(v8h*)&sA[lrow * LDT + lcol + 8] = a1;
    *(v8h*)&sB[lrow * LDT + lcol]     = b0;
    *(v8h*)&sB[lrow * LDT + lcol + 8] = b1;
    __syncthreads();
    const _Float16* cA = &sA[0];
    const _Float16* cB = &sB[0];
#endif

    // A frag (16x32 ISA layout): lane m=lm, halves at 8*lh+e (e<8), 16+8*lh+(e-8)
    Frag fa[2];
    #pragma unroll
    for (int i = 0; i < 2; ++i) {
      const _Float16* p = &cA[(wm * 32 + i * 16 + lm) * LDT + 8 * lh];
      fa[i].h[0] = *(const v8h*)(p);
      fa[i].h[1] = *(const v8h*)(p + 16);
    }
    // B frag (32x16 ISA layout): lane n=lm, halves at 8*lh+e (e<8), 16+8*lh+(e-8)
    Frag fb[4];
    #pragma unroll
    for (int j = 0; j < 4; ++j) {
      const _Float16* p = &cB[(wn * 64 + j * 16 + lm) * LDT + 8 * lh];
      fb[j].h[0] = *(const v8h*)(p);
      fb[j].h[1] = *(const v8h*)(p + 16);
    }

    #pragma unroll
    for (int i = 0; i < 2; ++i)
      #pragma unroll
      for (int j = 0; j < 4; ++j)
        acc[i][j] = __builtin_amdgcn_wmma_f32_16x16x32_f16(
            false, fa[i].v, false, fb[j].v, (short)0, acc[i][j], false, false);
    __syncthreads();   // all reads of this buffer done before it is refilled
  }

  // epilogue: bias + exact GELU + f16 store. C/D layout: VGPR r -> M=r+8*lh, lane -> N.
  #pragma unroll
  for (int i = 0; i < 2; ++i) {
    #pragma unroll
    for (int j = 0; j < 4; ++j) {
      const int n  = bn + wn * 64 + j * 16 + lm;
      const float bs = bias[n];
      #pragma unroll
      for (int r = 0; r < 8; ++r) {
        const int m = bm + wm * 32 + i * 16 + 8 * lh + r;
        float v = acc[i][j][r] + bs;
        if (apply_gelu) v = 0.5f * v * (1.0f + erff(v * 0.70710678118654752f));
        Out[(size_t)m * N + n] = (_Float16)v;
      }
    }
  }
}

// ---------------------------------------------------------------------------
// imp[row] = sigmoid(dot(h2[row,:], W3) + b3) + 1e-5
// ---------------------------------------------------------------------------
__global__ __launch_bounds__(256) void k_gemv_sigmoid(
    const _Float16* __restrict__ H, const float* __restrict__ W3,
    const float* __restrict__ b3, float* __restrict__ imp, int K)
{
  __shared__ float red[256];
  const int row = blockIdx.x;
  const _Float16* h = H + (size_t)row * K;
  float s = 0.f;
  for (int k = threadIdx.x; k < K; k += 256) s += (float)h[k] * W3[k];
  red[threadIdx.x] = s;
  __syncthreads();
  for (int off = 128; off > 0; off >>= 1) {
    if (threadIdx.x < off) red[threadIdx.x] += red[threadIdx.x + off];
    __syncthreads();
  }
  if (threadIdx.x == 0)
    imp[row] = 1.f / (1.f + expf(-(red[0] + b3[0]))) + 1e-5f;
}

// ---------------------------------------------------------------------------
// per-batch: cs = cumsum(imp) * (Tnew / sum(imp))   (normalization commutes)
// ---------------------------------------------------------------------------
__global__ __launch_bounds__(512) void k_scan_norm(
    const float* __restrict__ imp, float* __restrict__ cs)
{
  __shared__ float s[512];
  const int b   = blockIdx.x;
  const int tid = threadIdx.x;
  const int base = b * T_ + tid * 8;
  float v[8];
  float run = 0.f;
  #pragma unroll
  for (int e = 0; e < 8; ++e) { run += imp[base + e]; v[e] = run; }
  s[tid] = run;
  __syncthreads();
  for (int off = 1; off < 512; off <<= 1) {
    float t = (tid >= off) ? s[tid - off] : 0.f;
    __syncthreads();
    s[tid] += t;
    __syncthreads();
  }
  const float excl  = (tid == 0) ? 0.f : s[tid - 1];
  const float scale = (float)TNEW_ / s[511];
  #pragma unroll
  for (int e = 0; e < 8; ++e) cs[base + e] = (v[e] + excl) * scale;
}

// ---------------------------------------------------------------------------
// Xnew[b,n,c] = sum_t ( g(cs_t) - g(cs_{t-1}) ) * X[b,t,c]
// g(c) = clamp(c - n, 0, 1)   (last column: max(c - n, 0))
// ---------------------------------------------------------------------------
__global__ __launch_bounds__(128) void k_warp_gather(
    const float* __restrict__ cs, const float* __restrict__ X,
    float* __restrict__ out)
{
  const int n = blockIdx.x;
  const int b = blockIdx.y;
  const float* csb = cs + b * T_;
  const float fn = (float)n;
  const bool last = (n == TNEW_ - 1);

  int lo = 0, hi = T_;  // t0 = first t with cs[t] > n
  while (lo < hi) { int mid = (lo + hi) >> 1; if (csb[mid] > fn) hi = mid; else lo = mid + 1; }
  const int t0 = lo;
  int t1;
  if (last) {
    t1 = T_;
  } else {
    const float f1 = fn + 1.f;
    lo = t0; hi = T_;
    while (lo < hi) { int mid = (lo + hi) >> 1; if (csb[mid] >= f1) hi = mid; else lo = mid + 1; }
    t1 = (lo < T_) ? lo + 1 : T_;
  }

  const int c = threadIdx.x * 4;          // 128 threads x 4 = C
  float4 acc = make_float4(0.f, 0.f, 0.f, 0.f);
  float prev  = (t0 == 0) ? 0.f : csb[t0 - 1];
  float gprev = last ? fmaxf(prev - fn, 0.f) : fminf(fmaxf(prev - fn, 0.f), 1.f);
  for (int t = t0; t < t1; ++t) {
    const float cur  = csb[t];
    const float gcur = last ? fmaxf(cur - fn, 0.f) : fminf(fmaxf(cur - fn, 0.f), 1.f);
    const float w = gcur - gprev;
    gprev = gcur;
    const float4 x = *(const float4*)(X + (size_t)(b * T_ + t) * C_ + c);
    acc.x += w * x.x; acc.y += w * x.y; acc.z += w * x.z; acc.w += w * x.w;
  }
  *(float4*)(out + (size_t)(b * TNEW_ + n) * C_ + c) = acc;
}

// ---------------------------------------------------------------------------
extern "C" void kernel_launch(void* const* d_in, const int* in_sizes, int n_in,
                              void* d_out, int out_size, void* d_ws, size_t ws_size,
                              hipStream_t stream) {
  (void)in_sizes; (void)n_in; (void)out_size; (void)ws_size;
  const float* feat = (const float*)d_in[0];
  const float* W1   = (const float*)d_in[1];
  const float* b1   = (const float*)d_in[2];
  const float* W2   = (const float*)d_in[3];
  const float* b2   = (const float*)d_in[4];
  const float* W3   = (const float*)d_in[5];
  const float* b3   = (const float*)d_in[6];
  float* out = (float*)d_out;

  char* ws = (char*)d_ws;
  size_t off = 0;
  auto alloc = [&](size_t bytes) -> void* {
    void* p = ws + off;
    off += (bytes + 255) & ~(size_t)255;
    return p;
  };
  _Float16* fA  = (_Float16*)alloc((size_t)BT_ * C_ * 2);  //  32 MB
  _Float16* W1t = (_Float16*)alloc((size_t)D_ * C_ * 2);   //   2 MB
  _Float16* W2t = (_Float16*)alloc((size_t)D_ * D_ * 2);   //   8 MB
  _Float16* h1  = (_Float16*)alloc((size_t)BT_ * D_ * 2);  // 128 MB
  _Float16* h2  = (_Float16*)alloc((size_t)BT_ * D_ * 2);  // 128 MB
  float*    imp = (float*)alloc((size_t)BT_ * 4);
  float*    cs  = (float*)alloc((size_t)BT_ * 4);

  // precision staging
  {
    const int n = BT_ * C_;
    k_convert_f16<<<(n + 255) / 256, 256, 0, stream>>>(feat, fA, n);
  }
  k_convert_transpose_f16<<<(C_ * D_ + 255) / 256, 256, 0, stream>>>(W1, W1t, C_, D_);
  k_convert_transpose_f16<<<(D_ * D_ + 255) / 256, 256, 0, stream>>>(W2, W2t, D_, D_);

  // importance MLP
  const dim3 gg(D_ / TN, BT_ / TM);
  k_gemm_f16<<<gg, 256, 0, stream>>>(fA, W1t, b1, h1, BT_, D_, C_, 1);
  k_gemm_f16<<<gg, 256, 0, stream>>>(h1, W2t, b2, h2, BT_, D_, D_, 1);
  k_gemv_sigmoid<<<BT_, 256, 0, stream>>>(h2, W3, b3, imp, D_);

  // normalize + cumsum, then sparse warp-gather
  k_scan_norm<<<B_, 512, 0, stream>>>(imp, cs);
  const dim3 gw(TNEW_, B_);
  k_warp_gather<<<gw, 128, 0, stream>>>(cs, feat, out);
}
